// ErbDecoderStep_55765855371913
// MI455X (gfx1250) — compile-verified, matrix operand order checked by reading, not run on hardware
//
#include <hip/hip_runtime.h>
#include <hip/hip_bf16.h>

// ---------------------------------------------------------------------------
// CDNA5 (gfx1250) implementation of ErbDecoderStep.
//   kernel 1: fp32 -> bf16 weight conversion, pre-swizzled into WMMA B-frag order
//   kernel 2: grouped-linear + GRU step + grouped-linear via
//             v_wmma_f32_16x16x32_bf16 (fp32 accumulate).
//             32 batch rows / block (2 M-tiles) so every B fragment loaded
//             from L2 is reused twice -> halves L2 weight traffic vs 16 rows.
//   kernel 3: separable-conv decoder chain, fp32 VALU (small FLOP fraction)
// ---------------------------------------------------------------------------

typedef __attribute__((ext_vector_type(16))) __bf16 bf16x16;
typedef __attribute__((ext_vector_type(8)))  float  v8f;

union BFrag { bf16x16 v; unsigned u[8]; uint4 q[2]; };

__device__ __forceinline__ unsigned f2bf(float x) {
  unsigned u = __float_as_uint(x);
  return (u + 0x7fffu + ((u >> 16) & 1u)) >> 16;   // round-to-nearest-even
}
__device__ __forceinline__ float bf2f(unsigned short b) {
  return __uint_as_float(((unsigned)b) << 16);
}

// Per-lane K mapping for 16-bit A fragments of V_WMMA_*_16X16X32 (ISA 7.12.2):
//   lanes 0-15 : VGPR0-3 -> K=0..7,  VGPR4-7 -> K=16..23
//   lanes 16-31: VGPR0-3 -> K=8..15, VGPR4-7 -> K=24..31
// (B fragments use the mirrored layout with N = lane%16.)
__device__ __forceinline__ int kmap(int lane, int v) {
  int base = (lane < 16) ? 0 : 8;
  return (v < 4) ? (base + 2 * v) : (base + 8 + 2 * v);
}

// ---------------------------------------------------------------------------
// Workspace layout (d_ws):
//   fragment region: 792 fragments x 1024 B  (each frag: 32 lanes x 8 dwords)
//     frag [  0..383] : gru_k   (ntile 0..47, ktile 0..7)  f = ntile*8+ktile
//     frag [384..767] : gru_rk  (same layout, +384)
//     frag [768..783] : w_in    (g 0..7, ntile 0..1)       f = 768 + g*2+nt
//     frag [784..791] : w_out   (g 0..7)                   f = 784 + g
//   emb4 (fp32, B x 128) at byte offset 792*1024
// ---------------------------------------------------------------------------
#define FRAG_BYTES   (792 * 1024)

__global__ __launch_bounds__(256) void prep_weights_kernel(
    const float* __restrict__ w_in, const float* __restrict__ gru_k,
    const float* __restrict__ gru_rk, const float* __restrict__ w_out,
    unsigned* __restrict__ wf) {
  int f = blockIdx.x;                 // 0..791
  int t = threadIdx.x;                // 0..255
  int lane = t >> 3, v = t & 7;
  int k0 = kmap(lane, v);             // even k of the packed pair
  int n  = lane & 15;
  const float* src; int e0, stride;
  if (f < 384) {                      // gru_k: (256, 768)
    int nt = f >> 3, kt = f & 7;
    src = gru_k;  e0 = (kt * 32 + k0) * 768 + nt * 16 + n;  stride = 768;
  } else if (f < 768) {               // gru_rk
    int ff = f - 384, nt = ff >> 3, kt = ff & 7;
    src = gru_rk; e0 = (kt * 32 + k0) * 768 + nt * 16 + n;  stride = 768;
  } else if (f < 784) {               // w_in: (8, 32, 32)
    int ff = f - 768, g = ff >> 1, nt = ff & 1;
    src = w_in;   e0 = g * 1024 + k0 * 32 + nt * 16 + n;    stride = 32;
  } else {                            // w_out: (8, 32, 16)
    int g = f - 784;
    src = w_out;  e0 = g * 512 + k0 * 16 + n;               stride = 16;
  }
  unsigned lo = f2bf(src[e0]);
  unsigned hi = f2bf(src[e0 + stride]);
  wf[f * 256 + lane * 8 + v] = lo | (hi << 16);
}

// ---------------------------------------------------------------------------
// GRU core. block = 128 threads (4 wave32), 32 batch rows (2 M-tiles) / block.
// LDS rows padded to 264 halves (132 dwords, 132%64=4) -> conflict-free
// A-fragment ds reads (ds_load_b128 x2 per fragment).
// ---------------------------------------------------------------------------
#define XSTR 264

__global__ __launch_bounds__(128) void gru_gemm_kernel(
    const float* __restrict__ emb, const float* __restrict__ h_in,
    const float* __restrict__ gru_b, const unsigned* __restrict__ wf,
    float* __restrict__ emb4, float* __restrict__ out, int B) {
  __shared__ unsigned short emb16[32 * XSTR];
  __shared__ unsigned short h16[32 * XSTR];
  __shared__ unsigned short xin16[32 * XSTR];
  __shared__ unsigned short out16[32 * XSTR];

  const int tid   = threadIdx.x;
  const int bbase = blockIdx.x * 32;

  for (int i = tid; i < 32 * 256; i += 128) {
    int m = i >> 8, c = i & 255;
    emb16[m * XSTR + c] = (unsigned short)f2bf(emb[(size_t)(bbase + m) * 256 + c]);
    h16[m * XSTR + c]   = (unsigned short)f2bf(h_in[(size_t)(bbase + m) * 256 + c]);
  }
  __syncthreads();

  const int lane  = tid & 31;
  const int wv    = tid >> 5;          // wave id 0..3
  const int nloc  = lane & 15;         // D-matrix column within tile
  const int mrow  = lane & 15;         // A-matrix row
  const int mbase = (lane < 16) ? 0 : 8;
  int kof[8];
  #pragma unroll
  for (int v = 0; v < 8; ++v) kof[v] = kmap(lane, v);

  // ---- Phase A: x_in = relu(grouped_linear(emb, w_in)), B-frag reused 2x ----
  for (int i = 0; i < 4; ++i) {
    int t = wv * 4 + i, g = t >> 1, nt = t & 1;
    BFrag b;
    const uint4* bp = (const uint4*)(wf + (768 + t) * 256 + lane * 8);
    b.q[0] = bp[0]; b.q[1] = bp[1];
    int c = g * 32 + nt * 16 + nloc;
    #pragma unroll
    for (int mt = 0; mt < 2; ++mt) {
      BFrag a;
      int rb = (mt * 16 + mrow) * XSTR + g * 32;
      #pragma unroll
      for (int v = 0; v < 8; ++v)
        a.u[v] = *(const unsigned*)&emb16[rb + kof[v]];
      v8f acc = {};
      acc = __builtin_amdgcn_wmma_f32_16x16x32_bf16(false, a.v, false, b.v,
                                                    (short)0, acc, false, false);
      #pragma unroll
      for (int r = 0; r < 8; ++r) {
        float x = fmaxf(acc[r], 0.f);
        xin16[(mt * 16 + mbase + r) * XSTR + c] = (unsigned short)f2bf(x);
      }
    }
  }
  __syncthreads();

  // ---- Phase B: GRU gates. Each wave owns 64 hidden columns; each loaded
  //      B fragment feeds both M-tiles (12 WMMAs per fragment group). ----
  for (int jj = 0; jj < 4; ++jj) {
    int cb  = wv * 64 + jj * 16;       // hidden-state column base
    int ntz = cb >> 4;                 // z-gate N tile (0..15)
    // dword base offsets of the 6 fragment streams (z/r/n for gru_k, gru_rk)
    int fb[6];
    #pragma unroll
    for (int e = 0; e < 6; ++e)
      fb[e] = (((e >= 3) ? 384 : 0) + ((e % 3) * 16 + ntz) * 8) * 256 + lane * 8;

    v8f acc[12];
    #pragma unroll
    for (int q = 0; q < 12; ++q) acc[q] = (v8f){};

    for (int kt = 0; kt < 8; ++kt) {
      if (kt < 7) {
        #pragma unroll
        for (int e = 0; e < 6; ++e)
          __builtin_prefetch(wf + fb[e] + (kt + 1) * 256, 0, 3);
      }
      BFrag bf[6];
      #pragma unroll
      for (int e = 0; e < 6; ++e) {
        const uint4* p = (const uint4*)(wf + fb[e] + kt * 256);
        bf[e].q[0] = p[0]; bf[e].q[1] = p[1];
      }
      int kb = kt * 32;
      #pragma unroll
      for (int mt = 0; mt < 2; ++mt) {
        BFrag aX, aH;
        int rb = (mt * 16 + mrow) * XSTR + kb;
        #pragma unroll
        for (int v = 0; v < 8; ++v) {
          aX.u[v] = *(const unsigned*)&xin16[rb + kof[v]];
          aH.u[v] = *(const unsigned*)&h16[rb + kof[v]];
        }
        acc[mt*6+0] = __builtin_amdgcn_wmma_f32_16x16x32_bf16(false, aX.v, false, bf[0].v, (short)0, acc[mt*6+0], false, false);
        acc[mt*6+1] = __builtin_amdgcn_wmma_f32_16x16x32_bf16(false, aX.v, false, bf[1].v, (short)0, acc[mt*6+1], false, false);
        acc[mt*6+2] = __builtin_amdgcn_wmma_f32_16x16x32_bf16(false, aX.v, false, bf[2].v, (short)0, acc[mt*6+2], false, false);
        acc[mt*6+3] = __builtin_amdgcn_wmma_f32_16x16x32_bf16(false, aH.v, false, bf[3].v, (short)0, acc[mt*6+3], false, false);
        acc[mt*6+4] = __builtin_amdgcn_wmma_f32_16x16x32_bf16(false, aH.v, false, bf[4].v, (short)0, acc[mt*6+4], false, false);
        acc[mt*6+5] = __builtin_amdgcn_wmma_f32_16x16x32_bf16(false, aH.v, false, bf[5].v, (short)0, acc[mt*6+5], false, false);
      }
    }
    int c = cb + nloc;
    float b0z = gru_b[c],        b0r = gru_b[256 + c],  b0n = gru_b[512 + c];
    float b1z = gru_b[768 + c],  b1r = gru_b[1024 + c], b1n = gru_b[1280 + c];
    #pragma unroll
    for (int mt = 0; mt < 2; ++mt) {
      #pragma unroll
      for (int r = 0; r < 8; ++r) {
        int m = mt * 16 + mbase + r;
        size_t b = (size_t)(bbase + m);
        float xz = acc[mt*6+0][r] + b0z, xr = acc[mt*6+1][r] + b0r, xn = acc[mt*6+2][r] + b0n;
        float hz = acc[mt*6+3][r] + b1z, hr = acc[mt*6+4][r] + b1r, hn = acc[mt*6+5][r] + b1n;
        float zg = 1.f / (1.f + __expf(-(xz + hz)));
        float rg = 1.f / (1.f + __expf(-(xr + hr)));
        float ng = tanhf(xn + rg * hn);
        float hold = h_in[b * 256 + c];
        float hnew = zg * hold + (1.f - zg) * ng;
        out[(size_t)B * 32 + b * 256 + c] = hnew;               // new_state
        float xv = bf2f(xin16[m * XSTR + c]);
        out16[m * XSTR + c] = (unsigned short)f2bf(hnew + xv);  // h' + x_in
      }
    }
  }
  __syncthreads();

  // ---- Phase C: emb4 = relu(grouped_linear(h'+x_in, w_out)), B-frag x2 ----
  for (int i = 0; i < 2; ++i) {
    int g = wv * 2 + i;
    BFrag b;
    const uint4* bp = (const uint4*)(wf + (784 + g) * 256 + lane * 8);
    b.q[0] = bp[0]; b.q[1] = bp[1];
    int c = g * 16 + nloc;
    #pragma unroll
    for (int mt = 0; mt < 2; ++mt) {
      BFrag a;
      int rb = (mt * 16 + mrow) * XSTR + g * 32;
      #pragma unroll
      for (int v = 0; v < 8; ++v)
        a.u[v] = *(const unsigned*)&out16[rb + kof[v]];
      v8f acc = {};
      acc = __builtin_amdgcn_wmma_f32_16x16x32_bf16(false, a.v, false, b.v,
                                                    (short)0, acc, false, false);
      #pragma unroll
      for (int r = 0; r < 8; ++r)
        emb4[(size_t)(bbase + mt * 16 + mbase + r) * 128 + c] = fmaxf(acc[r], 0.f);
    }
  }
}

// ---------------------------------------------------------------------------
// Conv decoder (fp32 VALU). 16 rows / block, 256 threads, SA/SB LDS ping-pong.
// LDS layout: (row, f, c) -> row*512 + f*16 + c   (f stride fixed at 32)
// ---------------------------------------------------------------------------
struct ConvParams {
  const float* dw[7]; const float* pw[7]; const float* cg[7]; const float* cb[7];
  const float* out_w; const float* out_g; const float* out_b;
};

// kf=1 'VALID' separable block reading straight from global (gsrc already
// offset to this block's rows), optional LDS and global residuals.
__device__ void sep1(const float* __restrict__ gsrc, int F,
                     const float* __restrict__ dw, const float* __restrict__ pw,
                     const float* __restrict__ gg, const float* __restrict__ bb,
                     const float* __restrict__ addl, const float* __restrict__ addg,
                     float* __restrict__ dst, int tid) {
  int n = 16 * F * 16;
  for (int i = tid; i < n; i += 256) {
    int row = i / (F * 16), rem = i - row * (F * 16);
    int f = rem >> 4, o = rem & 15;
    float s = 0.f;
    #pragma unroll
    for (int c = 0; c < 16; ++c)
      s += gsrc[(size_t)row * F * 16 + f * 16 + c] * dw[c] * pw[c * 16 + o];
    float v = fmaxf(s * gg[o] + bb[o], 0.f);
    if (addl) v += addl[row * 512 + f * 16 + o];
    if (addg) v += addg[(size_t)row * 128 + rem];
    dst[row * 512 + f * 16 + o] = v;
  }
}

// kf=3, stride 1, same pad, no dilation.
__device__ void sep3(const float* __restrict__ src, float* __restrict__ dst,
                     const float* __restrict__ dw, const float* __restrict__ pw,
                     const float* __restrict__ gg, const float* __restrict__ bb,
                     int F, int tid) {
  int n = 16 * F * 16;
  for (int i = tid; i < n; i += 256) {
    int row = i / (F * 16), rem = i - row * (F * 16);
    int f = rem >> 4, o = rem & 15;
    float s = 0.f;
    #pragma unroll
    for (int c = 0; c < 16; ++c) {
      float y = 0.f;
      #pragma unroll
      for (int t = 0; t < 3; ++t) {
        int j = f + t - 1;
        if (j >= 0 && j < F) y += src[row * 512 + j * 16 + c] * dw[t * 16 + c];
      }
      s += y * pw[c * 16 + o];
    }
    dst[row * 512 + f * 16 + o] = fmaxf(s * gg[o] + bb[o], 0.f);
  }
}

// kf=3 transposed conv: lhs_dilation 2, pad (1,2) -> Fout = 2*Fin.
__device__ void sep3t(const float* __restrict__ src, float* __restrict__ dst,
                      const float* __restrict__ dw, const float* __restrict__ pw,
                      const float* __restrict__ gg, const float* __restrict__ bb,
                      int Fin, int tid) {
  int Fo = Fin * 2;
  int n = 16 * Fo * 16;
  for (int i = tid; i < n; i += 256) {
    int row = i / (Fo * 16), rem = i - row * (Fo * 16);
    int fo = rem >> 4, o = rem & 15;
    float s = 0.f;
    #pragma unroll
    for (int c = 0; c < 16; ++c) {
      float y = 0.f;
      #pragma unroll
      for (int t = 0; t < 3; ++t) {
        int j = fo + t - 1;                         // index into dilated input
        if (j >= 0 && j <= 2 * Fin - 2 && ((j & 1) == 0))
          y += src[row * 512 + (j >> 1) * 16 + c] * dw[t * 16 + c];
      }
      s += y * pw[c * 16 + o];
    }
    dst[row * 512 + fo * 16 + o] = fmaxf(s * gg[o] + bb[o], 0.f);
  }
}

__global__ __launch_bounds__(256) void dec_conv_kernel(
    const float* __restrict__ e3, const float* __restrict__ e2,
    const float* __restrict__ e1, const float* __restrict__ e0,
    const float* __restrict__ emb4, ConvParams p,
    float* __restrict__ out, int B) {
  __shared__ float SA[16 * 32 * 16];
  __shared__ float SB[16 * 32 * 16];
  const int tid   = threadIdx.x;
  const int bbase = blockIdx.x * 16;

  // x3 = c3p(e3) + emb4
  sep1(e3 + (size_t)bbase * 128, 8, p.dw[0], p.pw[0], p.cg[0], p.cb[0],
       nullptr, emb4 + (size_t)bbase * 128, SA, tid);
  __syncthreads();
  sep3(SA, SB, p.dw[1], p.pw[1], p.cg[1], p.cb[1], 8, tid);        // t3
  __syncthreads();
  sep1(e2 + (size_t)bbase * 128, 8, p.dw[2], p.pw[2], p.cg[2], p.cb[2],
       SB, nullptr, SA, tid);                                       // c2p + x3
  __syncthreads();
  sep3t(SA, SB, p.dw[3], p.pw[3], p.cg[3], p.cb[3], 8, tid);       // t2: 8->16
  __syncthreads();
  sep1(e1 + (size_t)bbase * 256, 16, p.dw[4], p.pw[4], p.cg[4], p.cb[4],
       SB, nullptr, SA, tid);                                       // c1p + x2
  __syncthreads();
  sep3t(SA, SB, p.dw[5], p.pw[5], p.cg[5], p.cb[5], 16, tid);      // t1: 16->32
  __syncthreads();
  sep1(e0 + (size_t)bbase * 512, 32, p.dw[6], p.pw[6], p.cg[6], p.cb[6],
       SB, nullptr, SA, tid);                                       // c0p + x1
  __syncthreads();

  // m = sigmoid(conv(x0, out_w, same) * g + b)
  const float* ow = p.out_w;
  float og = p.out_g[0], ob = p.out_b[0];
  for (int i = tid; i < 16 * 32; i += 256) {
    int row = i >> 5, f = i & 31;
    float s = 0.f;
    #pragma unroll
    for (int t = 0; t < 3; ++t) {
      int j = f + t - 1;
      if (j >= 0 && j < 32) {
        #pragma unroll
        for (int c = 0; c < 16; ++c)
          s += SA[row * 512 + j * 16 + c] * ow[t * 16 + c];
      }
    }
    float v = s * og + ob;
    out[(size_t)(bbase + row) * 32 + f] = 1.f / (1.f + __expf(-v));
  }
}

// ---------------------------------------------------------------------------
extern "C" void kernel_launch(void* const* d_in, const int* in_sizes, int n_in,
                              void* d_out, int out_size, void* d_ws, size_t ws_size,
                              hipStream_t stream) {
  const float* emb = (const float*)d_in[0];
  const float* e3  = (const float*)d_in[1];
  const float* e2  = (const float*)d_in[2];
  const float* e1  = (const float*)d_in[3];
  const float* e0  = (const float*)d_in[4];
  const float* hst = (const float*)d_in[5];
  // params flattened in setup_inputs() dict order:
  //   0 w_in, 1 gru_k, 2 gru_rk, 3 gru_b, 4 w_out,
  //   5+4i..8+4i: {dw,pw,g,b} for [c3p,t3,c2p,t2,c1p,t1,c0p],
  //   33 out_w, 34 out_g, 35 out_b
  const float* const* P = (const float* const*)(d_in + 6);

  int B = in_sizes[0] / 256;                    // 16384

  unsigned* wf   = (unsigned*)d_ws;
  float*    emb4 = (float*)((char*)d_ws + FRAG_BYTES);

  prep_weights_kernel<<<792, 256, 0, stream>>>(P[0], P[1], P[2], P[4], wf);
  gru_gemm_kernel<<<B / 32, 128, 0, stream>>>(emb, hst, P[3], wf, emb4,
                                              (float*)d_out, B);
  ConvParams cp;
  for (int i = 0; i < 7; ++i) {
    cp.dw[i] = P[5 + 4 * i]; cp.pw[i] = P[6 + 4 * i];
    cp.cg[i] = P[7 + 4 * i]; cp.cb[i] = P[8 + 4 * i];
  }
  cp.out_w = P[33]; cp.out_g = P[34]; cp.out_b = P[35];
  dec_conv_kernel<<<B / 16, 256, 0, stream>>>(e3, e2, e1, e0, emb4, cp,
                                              (float*)d_out, B);
}
